// LSTM__43688407335068
// MI455X (gfx1250) — matrix operand order repaired and checked by reading, not compiled, not run on hardware
//
#include <hip/hip_runtime.h>
#include <hip/hip_bf16.h>

// ---------------------------------------------------------------------------
// LSTM on MI455X (gfx1250, wave32, WMMA).
//  Phase 1: x_proj = x@Wi + (bi+bh)   -- bf16 WMMA GEMM, 32768x2048x512
//  Phase 2: 16 persistent workgroups (one per WGP), each owning 128 gate
//           columns with its Wh slice resident in LDS (128KB of the 320KB
//           CDNA5 LDS). Per-step cross-WGP h broadcast via L2 + atomic
//           barrier (+ cluster barrier, NOP outside cluster dispatch).
//           Next-step x_proj slice prefetched under the barrier spin.
// ---------------------------------------------------------------------------

typedef __attribute__((ext_vector_type(16))) __bf16    v16bf;
typedef __attribute__((ext_vector_type(8)))  float     v8f;
typedef __attribute__((ext_vector_type(4)))  unsigned  u32x4;

union FragBF {
    v16bf v;
    u32x4 q[2];
};

#define B_SZ   32
#define T_SZ   1024
#define I_SZ   512
#define H_SZ   512
#define N4H    2048                 // 4*H
#define M_ALL  (B_SZ * T_SZ)        // 32768
#define NBLK   16                   // persistent workgroups in phase 2

// -------------------------------------------------------------------- convert
// x(f32)->bf16 row-major; Wi,Wh -> transposed bf16 [N][K]; bias = bi+bh.
__global__ void lstm_cvt(const float* __restrict__ x,
                         const float* __restrict__ Wi,
                         const float* __restrict__ Wh,
                         const float* __restrict__ bi,
                         const float* __restrict__ bh,
                         __hip_bfloat16* __restrict__ xb,
                         __hip_bfloat16* __restrict__ WiT,
                         __hip_bfloat16* __restrict__ WhT,
                         float* __restrict__ bias) {
    const long NX = (long)M_ALL * I_SZ;          // 16,777,216
    const long NW = (long)I_SZ * N4H;            //  1,048,576
    const long TOT = NX + 2 * NW + N4H;
    for (long idx = (long)blockIdx.x * blockDim.x + threadIdx.x; idx < TOT;
         idx += (long)gridDim.x * blockDim.x) {
        if (idx < NX) {
            xb[idx] = __float2bfloat16(x[idx]);
        } else if (idx < NX + NW) {
            long j = idx - NX; long n = j / I_SZ, k = j % I_SZ;
            WiT[j] = __float2bfloat16(Wi[k * N4H + n]);
        } else if (idx < NX + 2 * NW) {
            long j = idx - NX - NW; long n = j / H_SZ, k = j % H_SZ;
            WhT[j] = __float2bfloat16(Wh[k * N4H + n]);
        } else {
            long n = idx - NX - 2 * NW;
            bias[n] = bi[n] + bh[n];
        }
    }
}

// --------------------------------------------------------------- phase 1 GEMM
// Workgroup = 8 waves, tile 128(m) x 64(n); each wave: 16(m) x 64(n), 4 accs.
__global__ void __launch_bounds__(256)
lstm_gemm_xproj(const __hip_bfloat16* __restrict__ xb,      // [M_ALL][512]
                const __hip_bfloat16* __restrict__ WiT,     // [2048][512]
                const float* __restrict__ bias,             // [2048]
                float* __restrict__ xproj) {                // [M_ALL][2048]
    const int lane = threadIdx.x & 31;
    const int wave = threadIdx.x >> 5;       // 0..7
    const int half = lane >> 4;
    const int l16  = lane & 15;
    const int mbase = blockIdx.x * 128 + wave * 16;
    const int nbase = blockIdx.y * 64;
    const int mrowA = mbase + l16;

    v8f acc[4];
#pragma unroll
    for (int j = 0; j < 4; ++j) {
        float bv = bias[nbase + j * 16 + l16];
#pragma unroll
        for (int r = 0; r < 8; ++r) acc[j][r] = bv;
    }

    const __hip_bfloat16* arow = xb + (long)mrowA * I_SZ;
#pragma unroll 4
    for (int kt = 0; kt < I_SZ / 32; ++kt) {
        FragBF a;
        // A 16x32 bf16: half 0 -> K 0..7 & 16..23 ; half 1 -> K 8..15 & 24..31
        a.q[0] = *(const u32x4*)(arow + kt * 32 + half * 8);
        a.q[1] = *(const u32x4*)(arow + kt * 32 + 16 + half * 8);
#pragma unroll
        for (int j = 0; j < 4; ++j) {
            // B 32x16 bf16 via WiT row n: half 0 -> K 0..7 & 16..23 ; half 1 -> K 8..15 & 24..31
            const __hip_bfloat16* brow =
                WiT + (long)(nbase + j * 16 + l16) * I_SZ + kt * 32 + half * 8;
            FragBF b;
            b.q[0] = *(const u32x4*)(brow);
            b.q[1] = *(const u32x4*)(brow + 16);
            acc[j] = __builtin_amdgcn_wmma_f32_16x16x32_bf16(
                false, a.v, false, b.v, (short)0, acc[j], false, false);
        }
    }
#pragma unroll
    for (int j = 0; j < 4; ++j) {
        int n = nbase + j * 16 + l16;
#pragma unroll
        for (int r = 0; r < 8; ++r) {
            int mr = mbase + r + half * 8;        // C/D layout: VGPR r -> M=r (+8)
            xproj[(long)mr * N4H + n] = acc[j][r];
        }
    }
}

__device__ __forceinline__ float sigmoidf_(float z) {
    return 1.0f / (1.0f + __expf(-z));
}

// ------------------------------------------------------------------ barrier
__global__ void lstm_barrier_init(unsigned* bar) {
    if (threadIdx.x == 0) *bar = 0u;
}

// ------------------------------------------------------------ phase 2 recur
// 16 persistent workgroups x 256 threads (8 waves). Workgroup g owns, for
// each gate q, h-columns [g*32, g*32+32): 8 n-tiles total, Wh slice in LDS.
// LDS: whs[128][512] bf16 (128KB) + hs[32][512] bf16 (32KB) + zs[32][128]
// f32 (16KB) = 176KB of the WGP's 320KB.
__global__ void __launch_bounds__(256)
lstm_recur(const __hip_bfloat16* __restrict__ WhT,   // [2048][512] bf16
           const float* __restrict__ xproj,          // [B*T][2048]
           __hip_bfloat16* __restrict__ hglob,       // [32][512] bf16 broadcast
           unsigned* __restrict__ bar,
           float* __restrict__ out,                  // [B][T][H]
           float* __restrict__ hT,                   // [B][H]
           float* __restrict__ cT) {                 // [B][H]
    extern __shared__ char smem[];
    __hip_bfloat16* whs = (__hip_bfloat16*)smem;                       // [128][512]
    __hip_bfloat16* hs  = (__hip_bfloat16*)(smem + 128 * H_SZ * 2);    // [32][512]
    float*          zs  = (float*)(smem + 128 * H_SZ * 2 + B_SZ * H_SZ * 2); // [32][128]

    const int g    = blockIdx.x;      // 0..15
    const int tid  = threadIdx.x;
    const int lane = tid & 31;
    const int wave = tid >> 5;        // 0..7 == local n-tile jt
    const int half = lane >> 4;
    const int l16  = lane & 15;
    const int jt   = wave;
    const int qg   = jt >> 1;         // gate 0..3
    const int eg   = jt & 1;
    const int gn_base = qg * H_SZ + g * 32 + eg * 16;   // global column base

    // ---- preload Wh slice into LDS: local row r=jt*16+rl -> WhT row gn ----
    for (int v = tid; v < 128 * H_SZ / 8; v += 256) {   // 8192 x (8 bf16)
        const int r  = v >> 6;           // local row 0..127
        const int kv = v & 63;           // 16B vector within row
        const int rjt = r >> 4, rl = r & 15;
        const int gn = (rjt >> 1) * H_SZ + g * 32 + (rjt & 1) * 16 + rl;
        ((u32x4*)(whs + r * H_SZ))[kv] = ((const u32x4*)(WhT + (long)gn * H_SZ))[kv];
    }
    // ---- h = 0 -------------------------------------------------------------
    const u32x4 zero4 = {0u, 0u, 0u, 0u};
    for (int v = tid; v < B_SZ * H_SZ / 8; v += 256) ((u32x4*)hs)[v] = zero4;

    // element-wise ownership: batch row eb, 4 local h-columns
    const int eb  = tid >> 3;         // 0..31
    const int eh0 = (tid & 7) * 4;    // 0..28
    float c[4], hreg[4];
#pragma unroll
    for (int j = 0; j < 4; ++j) { c[j] = 0.0f; hreg[j] = 0.0f; }
    __syncthreads();

    for (int t = 0; t < T_SZ; ++t) {
        // ---- z = xproj[t] + h @ Wh for this WGP's columns ------------------
        v8f acc[2];
#pragma unroll
        for (int mi = 0; mi < 2; ++mi)
#pragma unroll
            for (int r = 0; r < 8; ++r) {
                const int b = mi * 16 + r + half * 8;
                acc[mi][r] = xproj[((long)b * T_SZ + t) * N4H + gn_base + l16];
            }
        for (int kt = 0; kt < H_SZ / 32; ++kt) {
            FragBF a[2];
#pragma unroll
            for (int mi = 0; mi < 2; ++mi) {
                const __hip_bfloat16* ar =
                    hs + (mi * 16 + l16) * H_SZ + kt * 32 + half * 8;
                a[mi].q[0] = *(const u32x4*)(ar);
                a[mi].q[1] = *(const u32x4*)(ar + 16);
            }
            const __hip_bfloat16* br =
                whs + (jt * 16 + l16) * H_SZ + kt * 32 + half * 8;
            FragBF b;
            b.q[0] = *(const u32x4*)(br);
            b.q[1] = *(const u32x4*)(br + 16);
#pragma unroll
            for (int mi = 0; mi < 2; ++mi)
                acc[mi] = __builtin_amdgcn_wmma_f32_16x16x32_bf16(
                    false, a[mi].v, false, b.v, (short)0, acc[mi], false, false);
        }
#pragma unroll
        for (int mi = 0; mi < 2; ++mi)
#pragma unroll
            for (int r = 0; r < 8; ++r) {
                const int b = mi * 16 + r + half * 8;
                zs[b * 128 + jt * 16 + l16] = acc[mi][r];
            }
        __syncthreads();

        // ---- gates + state update (local columns) --------------------------
#pragma unroll
        for (int j = 0; j < 4; ++j) {
            const int hcl = eh0 + j;                         // 0..31
            const int ce  = (hcl >> 4) * 16 + (hcl & 15);
            const float zi = zs[eb * 128 + 0 * 32 + ce];
            const float zf = zs[eb * 128 + 1 * 32 + ce];
            const float zg = zs[eb * 128 + 2 * 32 + ce];
            const float zo = zs[eb * 128 + 3 * 32 + ce];
            const float ig = sigmoidf_(zi);
            const float fg = sigmoidf_(zf);
            const float gg = tanhf(zg);
            const float og = sigmoidf_(zo);
            const float cn = fg * c[j] + ig * gg;
            const float hn = og * tanhf(cn);
            c[j] = cn;
            hreg[j] = hn;
            out[((long)eb * T_SZ + t) * H_SZ + g * 32 + hcl] = hn;
            hglob[eb * H_SZ + g * 32 + hcl] = __float2bfloat16(hn);
        }

        // ---- prefetch next step's x_proj slice under the barrier spin ------
        if (t + 1 < T_SZ) {
#pragma unroll
            for (int q = 0; q < 4; ++q)
                __builtin_prefetch(
                    &xproj[((long)eb * T_SZ + (t + 1)) * N4H + q * H_SZ +
                           g * 32 + eh0],
                    0, 3);
        }

        __threadfence();
        __syncthreads();

        // ---- device-wide step barrier across the 16 WGPs -------------------
        if (tid == 0) {
            __builtin_amdgcn_s_cluster_barrier();   // NOP unless cluster-dispatched
            __hip_atomic_fetch_add(bar, 1u, __ATOMIC_RELEASE,
                                   __HIP_MEMORY_SCOPE_AGENT);
            const unsigned tgt = (unsigned)NBLK * (unsigned)(t + 1);
            while (__hip_atomic_load(bar, __ATOMIC_ACQUIRE,
                                     __HIP_MEMORY_SCOPE_AGENT) < tgt)
                __builtin_amdgcn_s_sleep(1);
        }
        __syncthreads();
        __threadfence();

        // ---- re-broadcast full h into LDS ----------------------------------
        for (int v = tid; v < B_SZ * H_SZ / 8; v += 256)
            ((u32x4*)hs)[v] = ((const u32x4*)hglob)[v];
        __syncthreads();
    }

#pragma unroll
    for (int j = 0; j < 4; ++j) {
        const int hcl = eh0 + j;
        hT[eb * H_SZ + g * 32 + hcl] = hreg[j];
        cT[eb * H_SZ + g * 32 + hcl] = c[j];
    }
}

// ---------------------------------------------------------------------------
extern "C" void kernel_launch(void* const* d_in, const int* in_sizes, int n_in,
                              void* d_out, int out_size, void* d_ws, size_t ws_size,
                              hipStream_t stream) {
    const float* x  = (const float*)d_in[0];
    const float* Wi = (const float*)d_in[1];
    const float* Wh = (const float*)d_in[2];
    const float* bi = (const float*)d_in[3];
    const float* bh = (const float*)d_in[4];

    float* out = (float*)d_out;                       // [32][1024][512]
    float* hT  = out + (long)B_SZ * T_SZ * H_SZ;      // [32][512]
    float* cT  = hT + B_SZ * H_SZ;                    // [32][512]

    char* ws = (char*)d_ws;
    float* xproj = (float*)ws;                 ws += (size_t)M_ALL * N4H * 4;   // 256 MB
    __hip_bfloat16* xb  = (__hip_bfloat16*)ws; ws += (size_t)M_ALL * I_SZ * 2;  //  32 MB
    __hip_bfloat16* WiT = (__hip_bfloat16*)ws; ws += (size_t)N4H * I_SZ * 2;    //   2 MB
    __hip_bfloat16* WhT = (__hip_bfloat16*)ws; ws += (size_t)N4H * H_SZ * 2;    //   2 MB
    float* bias = (float*)ws;                  ws += (size_t)N4H * 4;           //   8 KB
    __hip_bfloat16* hglob = (__hip_bfloat16*)ws; ws += (size_t)B_SZ * H_SZ * 2; //  32 KB
    unsigned* bar = (unsigned*)ws;

    lstm_cvt<<<4096, 256, 0, stream>>>(x, Wi, Wh, bi, bh, xb, WiT, WhT, bias);

    dim3 g1(M_ALL / 128, N4H / 64);
    lstm_gemm_xproj<<<g1, 256, 0, stream>>>(xb, WiT, bias, xproj);

    lstm_barrier_init<<<1, 32, 0, stream>>>(bar);

    // LDS: 128KB Wh slice + 32KB h + 16KB z = 176KB per WGP
    const size_t shmem = (size_t)128 * H_SZ * 2 + (size_t)B_SZ * H_SZ * 2 +
                         (size_t)B_SZ * 128 * 4;
    lstm_recur<<<NBLK, 256, shmem, stream>>>(WhT, xproj, hglob, bar, out, hT, cT);
}
